// LongMIL_89283780149651
// MI455X (gfx1250) — matrix operand order repaired and checked by reading, not compiled, not run on hardware
//
#include <hip/hip_runtime.h>

#define NTOK   8192
#define DIM    384
#define GRIDW  300
#define TABW   (16 * GRIDW)          // 4800
#define QUANTV 896.0f                // 112*2*4
#define SLOPEV 0.00390625f           // 2^-8
#define EPSV   1e-5f
#define L2E    1.4426950408889634f
#define RSQRT_D 0.051031036307982884f // 384^-0.5

typedef __bf16 bf16;
typedef __attribute__((ext_vector_type(16))) __bf16 v16bf;
typedef __attribute__((ext_vector_type(8)))  __bf16 v8bf;
typedef __attribute__((ext_vector_type(8)))  float  v8f;
typedef __attribute__((ext_vector_type(4)))  int    v4i;

// ---- CDNA5 async global->LDS path (guarded; fallback = sync copy) ----------
#if defined(__has_builtin)
#if __has_builtin(__builtin_amdgcn_global_load_async_to_lds_b128)
#define LM_ASYNC 1
#endif
#endif
#ifndef LM_ASYNC
#define LM_ASYNC 0
#endif

#if LM_ASYNC
typedef __attribute__((address_space(1))) v4i as1_v4i;
typedef __attribute__((address_space(3))) v4i as3_v4i;
static __device__ __forceinline__ void lm_async_b128(const bf16* g, bf16* l) {
  __builtin_amdgcn_global_load_async_to_lds_b128((as1_v4i*)g, (as3_v4i*)l, 0, 0);
}
static __device__ __forceinline__ void lm_wait_async() {
#if __has_builtin(__builtin_amdgcn_s_wait_asynccnt)
  __builtin_amdgcn_s_wait_asynccnt(0);
#else
  asm volatile("s_wait_asynccnt 0" ::: "memory");
#endif
}
#else
static __device__ __forceinline__ void lm_wait_async() {}
#endif

static __device__ __forceinline__ v8f wmma_bf16(v16bf a, v16bf b, v8f c) {
  // D = A(16x32 bf16) * B(32x16 bf16) + C(16x16 f32)
  return __builtin_amdgcn_wmma_f32_16x16x32_bf16(false, a, false, b, (short)0, c,
                                                 false, false);
}

// A-operand fragment: lane holds K = {base..base+15} of its row.
// p points at &row[k_tile_base + 16*(lane>=16)]
static __device__ __forceinline__ v16bf ld_afrag(const bf16* p) {
  return *(const v16bf*)(p);
}
static __device__ __forceinline__ v8f zero8() {
  v8f z;
#pragma unroll
  for (int i = 0; i < 8; ++i) z[i] = 0.0f;
  return z;
}

// ---------------------------------------------------------------- small setup
__global__ void lm_init_kernel(int* mins, float* meanv) {
  int t = threadIdx.x;
  if (t < DIM) meanv[t] = 0.0f;
  if (t == 0) { mins[0] = 0x7f800000; mins[1] = 0x7f800000; }
}

__global__ void lm_minpos_kernel(const float* __restrict__ x, int* mins) {
  int i = blockIdx.x * blockDim.x + threadIdx.x;
  float px = x[(size_t)i * 386 + 384];
  float py = x[(size_t)i * 386 + 385];
  atomicMin(&mins[0], __float_as_int(px));   // values are non-negative
  atomicMin(&mins[1], __float_as_int(py));
}

__global__ void lm_idx_kernel(const float* __restrict__ x, const int* __restrict__ mins,
                              int* __restrict__ idx) {
  int i = blockIdx.x * blockDim.x + threadIdx.x;
  float mnx = __int_as_float(mins[0]), mny = __int_as_float(mins[1]);
  float px = x[(size_t)i * 386 + 384];
  float py = x[(size_t)i * 386 + 385];
  int xi = __float2int_rn((px - mnx) * (1.0f / QUANTV));
  int yi = __float2int_rn((py - mny) * (1.0f / QUANTV));
  idx[i] = xi * GRIDW + yi;
}

__global__ void lm_castw_kernel(const float* __restrict__ w, bf16* __restrict__ o, int n) {
  int i = blockIdx.x * blockDim.x + threadIdx.x;
  if (i < n) o[i] = (bf16)w[i];
}

// one wave per row, LN over 384, emit bf16
__global__ __launch_bounds__(256) void lm_ln_kernel(const float* __restrict__ in, int istride,
                                                    const float* __restrict__ g,
                                                    const float* __restrict__ b,
                                                    bf16* __restrict__ out) {
  int row  = blockIdx.x * 8 + (threadIdx.x >> 5);
  int lane = threadIdx.x & 31;
  const float* r = in + (size_t)row * istride;
  float vals[12], s = 0.0f, q = 0.0f;
#pragma unroll
  for (int i = 0; i < 12; ++i) {
    float v = r[lane + 32 * i];
    vals[i] = v; s += v; q += v * v;
  }
#pragma unroll
  for (int m = 16; m >= 1; m >>= 1) { s += __shfl_xor(s, m); q += __shfl_xor(q, m); }
  float mean = s * (1.0f / DIM);
  float var  = q * (1.0f / DIM) - mean * mean;
  float rstd = rsqrtf(var + EPSV);
  bf16* o = out + (size_t)row * DIM;
#pragma unroll
  for (int i = 0; i < 12; ++i) {
    int c = lane + 32 * i;
    o[c] = (bf16)((vals[i] - mean) * rstd * g[c] + b[c]);
  }
}

// ------------------------------------------------------------------ QKV GEMM
// out = xln(8192x384) @ w^T(384x1152); Q gets D^-0.5 folded in; bf16 outputs.
__global__ __launch_bounds__(256) void lm_gemm_qkv_kernel(const bf16* __restrict__ xln,
                                                          const bf16* __restrict__ w,
                                                          bf16* __restrict__ Qb,
                                                          bf16* __restrict__ Kb,
                                                          bf16* __restrict__ Vb) {
  const int wgl = blockIdx.x * 8 + (threadIdx.x >> 5);
  const int mt = wgl % (NTOK / 16);
  const int nt = wgl / (NTOK / 16);
  const int m0 = mt * 16, n0 = nt * 64;
  const int lane = threadIdx.x & 31, hi = lane >> 4, col = lane & 15;
  v8f acc[4];
#pragma unroll
  for (int t = 0; t < 4; ++t) acc[t] = zero8();
  for (int kk = 0; kk < DIM / 32; ++kk) {
    v16bf A = ld_afrag(xln + (size_t)(m0 + col) * DIM + kk * 32 + 16 * hi);
#pragma unroll
    for (int t = 0; t < 4; ++t) {
      v16bf B = *(const v16bf*)(w + (size_t)(n0 + 16 * t + col) * DIM + kk * 32 + 16 * hi);
      acc[t] = wmma_bf16(A, B, acc[t]);
    }
  }
#pragma unroll
  for (int t = 0; t < 4; ++t) {
    int n = n0 + 16 * t + col;
#pragma unroll
    for (int r = 0; r < 8; ++r) {
      int m = m0 + r + 8 * hi;
      float v = acc[t][r];
      if (n < DIM)          Qb[(size_t)m * DIM + n]           = (bf16)(v * RSQRT_D);
      else if (n < 2 * DIM) Kb[(size_t)m * DIM + (n - DIM)]   = (bf16)v;
      else                  Vb[(size_t)m * DIM + (n - 2*DIM)] = (bf16)v;
    }
  }
}

// ------------------------------------------------------------------ proj GEMM
// h = Ob(8192x384) @ w^T(384x384) + bias, f32 output.
__global__ __launch_bounds__(256) void lm_gemm_proj_kernel(const bf16* __restrict__ Ob,
                                                           const bf16* __restrict__ w,
                                                           const float* __restrict__ bias,
                                                           float* __restrict__ hout) {
  const int wgl = blockIdx.x * 8 + (threadIdx.x >> 5);
  const int mt = wgl % (NTOK / 16);
  const int nt = wgl / (NTOK / 16);
  const int m0 = mt * 16, n0 = nt * 64;
  const int lane = threadIdx.x & 31, hi = lane >> 4, col = lane & 15;
  v8f acc[4];
#pragma unroll
  for (int t = 0; t < 4; ++t) acc[t] = zero8();
  for (int kk = 0; kk < DIM / 32; ++kk) {
    v16bf A = ld_afrag(Ob + (size_t)(m0 + col) * DIM + kk * 32 + 16 * hi);
#pragma unroll
    for (int t = 0; t < 4; ++t) {
      v16bf B = *(const v16bf*)(w + (size_t)(n0 + 16 * t + col) * DIM + kk * 32 + 16 * hi);
      acc[t] = wmma_bf16(A, B, acc[t]);
    }
  }
#pragma unroll
  for (int t = 0; t < 4; ++t) {
    int n = n0 + 16 * t + col;
#pragma unroll
    for (int r = 0; r < 8; ++r) {
      int m = m0 + r + 8 * hi;
      hout[(size_t)m * DIM + n] = acc[t][r] + bias[n];
    }
  }
}

// ---------------------------------------------------------- flash attention
// 8 waves/WG; wave = 16 queries x full D. Double-buffered async global->LDS
// staging of K / V (ASYNCcnt); V transposed LDS->LDS for the P*V B-operand.
// Per 32-key block: 24 WMMAs for S, gathered ALiBi bias, online softmax (f32),
// 24 WMMAs for O += P*V.
__global__ __launch_bounds__(256) void lm_attn_kernel(const bf16* __restrict__ Qb,
                                                      const bf16* __restrict__ Kb,
                                                      const bf16* __restrict__ Vb,
                                                      const int* __restrict__ idx,
                                                      const float* __restrict__ table,
                                                      bf16* __restrict__ Ob) {
  extern __shared__ __attribute__((aligned(32))) char smem[];
  bf16* sQ   = (bf16*)smem;              // 8 * 16*384
  bf16* sK0  = sQ   + 8 * 16 * DIM;      // 32*384 row-major [key][d]
  bf16* sK1  = sK0  + 32 * DIM;
  bf16* sVr0 = sK1  + 32 * DIM;          // 32*384 row-major V staging
  bf16* sVr1 = sVr0 + 32 * DIM;
  bf16* sVT  = sVr1 + 32 * DIM;          // 384*32 transposed [d][key]
  bf16* sP   = sVT  + DIM * 32;          // 8 * 16*32
  const int tid = threadIdx.x;
  const int wid = tid >> 5, lane = tid & 31, hi = lane >> 4, col = lane & 15;
  const int q0 = (blockIdx.x * 8 + wid) * 16;
  bf16* sQw = sQ + wid * 16 * DIM;

  auto stage_kv = [&](int blk, int buf) {
    const bf16* gk = Kb + (size_t)blk * 32 * DIM;
    const bf16* gv = Vb + (size_t)blk * 32 * DIM;
    bf16* dk = buf ? sK1 : sK0;
    bf16* dv = buf ? sVr1 : sVr0;
#if LM_ASYNC
    for (int i = tid; i < 32 * DIM / 8; i += 256) {
      lm_async_b128(gk + (size_t)i * 8, dk + (size_t)i * 8);
      lm_async_b128(gv + (size_t)i * 8, dv + (size_t)i * 8);
    }
#else
    for (int i = tid; i < 32 * DIM / 8; i += 256) {
      ((v8bf*)dk)[i] = ((const v8bf*)gk)[i];
      ((v8bf*)dv)[i] = ((const v8bf*)gv)[i];
    }
    if (blk + 1 < NTOK / 32) {   // global_prefetch_b8 for the block after
      __builtin_prefetch(gk + 32 * DIM + tid * 48, 0, 1);
      __builtin_prefetch(gv + 32 * DIM + tid * 48, 0, 1);
    }
#endif
  };

  {  // stage this wave's Q tile (rows contiguous), D^-0.5 already folded in
    const bf16* src = Qb + (size_t)q0 * DIM;
#if LM_ASYNC
    for (int i = lane; i < 16 * DIM / 8; i += 32)
      lm_async_b128(src + (size_t)i * 8, sQw + (size_t)i * 8);
#else
    for (int i = lane; i < 16 * DIM / 8; i += 32)
      ((v8bf*)sQw)[i] = ((const v8bf*)src)[i];
#endif
  }
  stage_kv(0, 0);  // pre-issue key-block 0

  int iq[8];
#pragma unroll
  for (int r = 0; r < 8; ++r) iq[r] = idx[q0 + r + 8 * hi];
  float mrun[8], lrun[8];
#pragma unroll
  for (int r = 0; r < 8; ++r) { mrun[r] = -3.0e38f; lrun[r] = 0.0f; }
  v8f acc[24];
#pragma unroll
  for (int t = 0; t < 24; ++t) acc[t] = zero8();

  for (int ib = 0; ib < NTOK / 32; ++ib) {
    const int cur = ib & 1;
    const int kb = ib * 32;
    lm_wait_async();       // K/V (and Q on ib==0) for this block are in LDS
    __syncthreads();       // everyone done with previous buffers + data visible

    // transpose V: sVr[cur] (row-major) -> sVT [d][key]
    {
      const bf16* vr = cur ? sVr1 : sVr0;
      for (int i = tid; i < 32 * DIM / 8; i += 256) {
        v8bf v = *(const v8bf*)(vr + (size_t)i * 8);
        int key = i / (DIM / 8);
        int d0  = (i % (DIM / 8)) * 8;
#pragma unroll
        for (int j = 0; j < 8; ++j) sVT[(d0 + j) * 32 + key] = v[j];
      }
    }
    if (ib + 1 < NTOK / 32) stage_kv(ib + 1, cur ^ 1);  // overlap with compute
    __syncthreads();

    // S tiles: left keys kb..kb+15, right keys kb+16..kb+31
    const bf16* sKc = cur ? sK1 : sK0;
    v8f Sl = zero8(), Sr = zero8();
    for (int kk = 0; kk < DIM / 32; ++kk) {
      v16bf A  = ld_afrag(sQw + (size_t)col * DIM + kk * 32 + 16 * hi);
      v16bf Bl = *(const v16bf*)(sKc + (size_t)col * DIM + kk * 32 + 16 * hi);
      v16bf Br = *(const v16bf*)(sKc + (size_t)(16 + col) * DIM + kk * 32 + 16 * hi);
      Sl = wmma_bf16(A, Bl, Sl);
      Sr = wmma_bf16(A, Br, Sr);
    }
    // ALiBi bias gather (L2-resident table)
    const int ikl = idx[kb + col], ikr = idx[kb + 16 + col];
#pragma unroll
    for (int r = 0; r < 8; ++r) {
      Sl[r] += SLOPEV * table[(size_t)iq[r] * TABW + ikl];
      Sr[r] += SLOPEV * table[(size_t)iq[r] * TABW + ikr];
    }
    // online softmax: row stats replicated across the 16 lanes of each half
    float bm[8];
#pragma unroll
    for (int r = 0; r < 8; ++r) bm[r] = fmaxf(Sl[r], Sr[r]);
#pragma unroll
    for (int m = 1; m < 16; m <<= 1) {
#pragma unroll
      for (int r = 0; r < 8; ++r) bm[r] = fmaxf(bm[r], __shfl_xor(bm[r], m));
    }
    float pl[8], pr[8], rs[8], sc[8];
#pragma unroll
    for (int r = 0; r < 8; ++r) {
      float mn = fmaxf(mrun[r], bm[r]);
      sc[r] = exp2f((mrun[r] - mn) * L2E);
      mrun[r] = mn;
      pl[r] = exp2f((Sl[r] - mn) * L2E);
      pr[r] = exp2f((Sr[r] - mn) * L2E);
      rs[r] = pl[r] + pr[r];
    }
#pragma unroll
    for (int m = 1; m < 16; m <<= 1) {
#pragma unroll
      for (int r = 0; r < 8; ++r) rs[r] += __shfl_xor(rs[r], m);
    }
#pragma unroll
    for (int r = 0; r < 8; ++r) lrun[r] = lrun[r] * sc[r] + rs[r];
#pragma unroll
    for (int t = 0; t < 24; ++t) {
#pragma unroll
      for (int r = 0; r < 8; ++r) acc[t][r] *= sc[r];
    }
    // P (C-layout) -> LDS -> reload as A-operand layout for O += P * V
    bf16* pw = sP + wid * 16 * 32;
#pragma unroll
    for (int r = 0; r < 8; ++r) {
      pw[(r + 8 * hi) * 32 + col]      = (bf16)pl[r];
      pw[(r + 8 * hi) * 32 + 16 + col] = (bf16)pr[r];
    }
    asm volatile("s_wait_dscnt 0" ::: "memory");   // order in-wave LDS RAW
    v16bf Pa = ld_afrag(pw + col * 32 + 16 * hi);
#pragma unroll
    for (int t = 0; t < 24; ++t) {
      v16bf Bv = *(const v16bf*)(sVT + (size_t)(t * 16 + col) * 32 + 16 * hi);
      acc[t] = wmma_bf16(Pa, Bv, acc[t]);
    }
  }
  float inv[8];
#pragma unroll
  for (int r = 0; r < 8; ++r) inv[r] = 1.0f / lrun[r];
#pragma unroll
  for (int t = 0; t < 24; ++t) {
#pragma unroll
    for (int r = 0; r < 8; ++r) {
      Ob[(size_t)(q0 + r + 8 * hi) * DIM + t * 16 + col] = (bf16)(acc[t][r] * inv[r]);
    }
  }
}

// ---------------------------------------------------------------- head
__global__ __launch_bounds__(384) void lm_mean_kernel(const float* __restrict__ h,
                                                      float* __restrict__ meanv) {
  int col = threadIdx.x;
  int r0  = blockIdx.x * 128;
  float local = 0.0f;
  for (int k = 0; k < 128; ++k) local += h[(size_t)(r0 + k) * DIM + col];
  atomicAdd(&meanv[col], local);
}

__global__ __launch_bounds__(384) void lm_final_kernel(const float* __restrict__ meanv,
                                                       const float* __restrict__ g,
                                                       const float* __restrict__ b,
                                                       const float* __restrict__ w,
                                                       const float* __restrict__ bias,
                                                       float* __restrict__ out) {
  __shared__ float sv[DIM];
  __shared__ float stats[2];
  int t = threadIdx.x;
  float v = meanv[t] * (1.0f / NTOK);
  sv[t] = v;
  __syncthreads();
  if (t == 0) {
    float s = 0.0f, q = 0.0f;
    for (int i = 0; i < DIM; ++i) { s += sv[i]; q += sv[i] * sv[i]; }
    float m = s * (1.0f / DIM);
    stats[0] = m;
    stats[1] = rsqrtf(q * (1.0f / DIM) - m * m + EPSV);
  }
  __syncthreads();
  sv[t] = (v - stats[0]) * stats[1] * g[t] + b[t];
  __syncthreads();
  if (t == 0) {
    float l0 = bias[0], l1 = bias[1];
    for (int i = 0; i < DIM; ++i) { l0 += w[i] * sv[i]; l1 += w[DIM + i] * sv[i]; }
    out[0] = l0; out[1] = l1;
    out[2] = (l1 > l0) ? 1.0f : 0.0f;
    float mx = fmaxf(l0, l1);
    float e0 = expf(l0 - mx), e1 = expf(l1 - mx), s = e0 + e1;
    out[3] = e0 / s; out[4] = e1 / s;
  }
}

// ---------------------------------------------------------------- launch
extern "C" void kernel_launch(void* const* d_in, const int* in_sizes, int n_in,
                              void* d_out, int out_size, void* d_ws, size_t ws_size,
                              hipStream_t stream) {
  (void)in_sizes; (void)n_in; (void)out_size; (void)ws_size;
  const float* x       = (const float*)d_in[0];
  const float* table   = (const float*)d_in[1];
  const float* ln1_g   = (const float*)d_in[2];
  const float* ln1_b   = (const float*)d_in[3];
  const float* qkv1_w  = (const float*)d_in[4];
  const float* proj1_w = (const float*)d_in[5];
  const float* proj1_b = (const float*)d_in[6];
  const float* ln2_g   = (const float*)d_in[7];
  const float* ln2_b   = (const float*)d_in[8];
  const float* qkv2_w  = (const float*)d_in[9];
  const float* proj2_w = (const float*)d_in[10];
  const float* proj2_b = (const float*)d_in[11];
  const float* lnf_g   = (const float*)d_in[12];
  const float* lnf_b   = (const float*)d_in[13];
  const float* fc2_w   = (const float*)d_in[14];
  const float* fc2_b   = (const float*)d_in[15];

  char* ws = (char*)d_ws;
  size_t off = 0;
  auto carve = [&](size_t bytes) -> char* {
    char* p = ws + off;
    off = (off + bytes + 255) & ~(size_t)255;
    return p;
  };
  int*   idx   = (int*)carve((size_t)NTOK * 4);
  int*   mins  = (int*)carve(256);
  float* meanv = (float*)carve(DIM * 4);
  bf16*  xln   = (bf16*)carve((size_t)NTOK * DIM * 2);
  bf16*  Qb    = (bf16*)carve((size_t)NTOK * DIM * 2);
  bf16*  Kb    = (bf16*)carve((size_t)NTOK * DIM * 2);
  bf16*  Vb    = (bf16*)carve((size_t)NTOK * DIM * 2);
  bf16*  Obuf  = (bf16*)carve((size_t)NTOK * DIM * 2);
  float* h     = (float*)carve((size_t)NTOK * DIM * 4);
  bf16*  wqkv  = (bf16*)carve((size_t)3 * DIM * DIM * 2);
  bf16*  wproj = (bf16*)carve((size_t)DIM * DIM * 2);

  // sQ + 2*sK + 2*sVrow + sVT + sP
  const int attn_smem =
      (8 * 16 * DIM + 2 * 32 * DIM + 2 * 32 * DIM + DIM * 32 + 8 * 16 * 32) * 2;

  lm_init_kernel<<<1, 512, 0, stream>>>(mins, meanv);
  lm_minpos_kernel<<<NTOK / 256, 256, 0, stream>>>(x, mins);
  lm_idx_kernel<<<NTOK / 256, 256, 0, stream>>>(x, mins, idx);

  // ---- layer 1
  lm_castw_kernel<<<(3 * DIM * DIM) / 256, 256, 0, stream>>>(qkv1_w, wqkv, 3 * DIM * DIM);
  lm_castw_kernel<<<(DIM * DIM) / 256, 256, 0, stream>>>(proj1_w, wproj, DIM * DIM);
  lm_ln_kernel<<<NTOK / 8, 256, 0, stream>>>(x, 386, ln1_g, ln1_b, xln);
  lm_gemm_qkv_kernel<<<(NTOK / 16) * (3 * DIM / 64) / 8, 256, 0, stream>>>(xln, wqkv, Qb, Kb, Vb);
  lm_attn_kernel<<<NTOK / 128, 256, attn_smem, stream>>>(Qb, Kb, Vb, idx, table, Obuf);
  lm_gemm_proj_kernel<<<(NTOK / 16) * (DIM / 64) / 8, 256, 0, stream>>>(Obuf, wproj, proj1_b, h);

  // ---- layer 2
  lm_castw_kernel<<<(3 * DIM * DIM) / 256, 256, 0, stream>>>(qkv2_w, wqkv, 3 * DIM * DIM);
  lm_castw_kernel<<<(DIM * DIM) / 256, 256, 0, stream>>>(proj2_w, wproj, DIM * DIM);
  lm_ln_kernel<<<NTOK / 8, 256, 0, stream>>>(h, DIM, ln2_g, ln2_b, xln);
  lm_gemm_qkv_kernel<<<(NTOK / 16) * (3 * DIM / 64) / 8, 256, 0, stream>>>(xln, wqkv, Qb, Kb, Vb);
  lm_attn_kernel<<<NTOK / 128, 256, attn_smem, stream>>>(Qb, Kb, Vb, idx, table, Obuf);
  lm_gemm_proj_kernel<<<(NTOK / 16) * (DIM / 64) / 8, 256, 0, stream>>>(Obuf, wproj, proj2_b, h);

  // ---- head
  lm_mean_kernel<<<NTOK / 128, 384, 0, stream>>>(h, meanv);
  lm_final_kernel<<<1, 384, 0, stream>>>(meanv, lnf_g, lnf_b, fc2_w, fc2_b, (float*)d_out);
}
